// LinearAttention_48112223650146
// MI455X (gfx1250) — matrix operand order repaired and checked by reading, not compiled, not run on hardware
//
#include <hip/hip_runtime.h>
#include <hip/hip_bf16.h>

// ---------------------------------------------------------------------------
// Linear attention, reassociated: out = (v @ (k_softmaxed^T @ q_softmaxed)) W_o
// All GEMMs on v_wmma_f32_16x16x32_bf16 (CDNA5), f32 accumulate.
// Big GEMMs: block-cooperative LDS staging of the B panel via async-to-LDS
// (ASYNCcnt) with double buffering; B fragments via ds_load_b128.
// ---------------------------------------------------------------------------

#define BATCH 2
#define SEQ   2048
#define DIM   512
#define HEADS 8
#define DH    64
#define INNER 512
#define QKV3  1536
#define ROWS  (BATCH * SEQ)   // 4096
#define KC    128             // K-chunk staged in LDS
#define LDSPAD 8              // bf16 elems of row padding (16B) -> bank spread

typedef __attribute__((ext_vector_type(16))) __bf16 v16bf;
typedef __attribute__((ext_vector_type(8)))  float  v8f;
typedef __attribute__((ext_vector_type(4)))  int    v4i;

#define GLOBAL_AS __attribute__((address_space(1)))
#define LOCAL_AS  __attribute__((address_space(3)))

#if __has_builtin(__builtin_amdgcn_global_load_async_to_lds_b128)
#define HAVE_ASYNC_LDS 1
#define ASYNC_COPY16(g, l) \
    __builtin_amdgcn_global_load_async_to_lds_b128((GLOBAL_AS v4i*)(g), \
                                                   (LOCAL_AS v4i*)(l), 0, 0)
#else
#define HAVE_ASYNC_LDS 0
#define ASYNC_COPY16(g, l) (*(uint4*)(l) = *(const uint4*)(g))
#endif

__device__ inline void wait_async_copies() {
#if __has_builtin(__builtin_amdgcn_s_wait_asynccnt)
    __builtin_amdgcn_s_wait_asynccnt(0);
#elif HAVE_ASYNC_LDS
    asm volatile("s_wait_asynccnt 0x0" ::: "memory");
#endif
}

// ---------------- WMMA fragment helpers (contiguous k = 16*(lane/16) + s for A and B, not the ISA table order) ----

__device__ inline v16bf load_frag_a(const __bf16* A, int lda, int row0, int k0, int lane) {
    // 16x32 bf16 A tile: lane m (mod 16) holds row m; low half-wave K=+0..15,
    // high half-wave K=+16..31, a 16-element contiguous run (32B).
    int m  = lane & 15;
    int hi = (lane >> 4) & 1;
    const __bf16* p = A + (size_t)(row0 + m) * lda + (k0 + hi * 16);
    union { v16bf v; uint4 q[2]; } u;
    u.q[0] = *(const uint4*)(p);        // elems 0..7  : K = k0 + hi*16 + 0..7
    u.q[1] = *(const uint4*)(p + 8);    // elems 8..15 : K = k0 + hi*16 + 8..15
    return u.v;
}

__device__ inline v16bf load_frag_b(const __bf16* Bw, int ldb, int col0, int k0, int lane) {
    // 32x16 bf16 B tile, B in weight layout (N x K row-major):
    // lane n holds column n = contiguous K-run of 16 elements (32B).
    int n  = lane & 15;
    int hi = (lane >> 4) & 1;
    const __bf16* p = Bw + (size_t)(col0 + n) * ldb + (k0 + hi * 16);
    union { v16bf v; uint4 q[2]; } u;
    u.q[0] = *(const uint4*)(p);
    u.q[1] = *(const uint4*)(p + 8);
    return u.v;
}

__device__ inline v8f wmma_bf16(v16bf a, v16bf b, v8f c) {
    // (neg_a, A, neg_b, B, c_mod, C, reuse_a, reuse_b)
    return __builtin_amdgcn_wmma_f32_16x16x32_bf16(false, a, false, b, (short)0, c,
                                                   false, false);
}

// ---------------- small reductions (wave32) --------------------------------

__device__ inline float wave_max(float v) {
    #pragma unroll
    for (int o = 16; o > 0; o >>= 1) v = fmaxf(v, __shfl_xor(v, o, 32));
    return v;
}
__device__ inline float wave_sum(float v) {
    #pragma unroll
    for (int o = 16; o > 0; o >>= 1) v += __shfl_xor(v, o, 32);
    return v;
}

// ---------------- kernels --------------------------------------------------

__global__ void cvt_f32_bf16(const float* __restrict__ in, __bf16* __restrict__ out, int n) {
    int i = blockIdx.x * blockDim.x + threadIdx.x;
    if (i < n) out[i] = (__bf16)in[i];
}

// C(MxN,f32) = A(MxK bf16 row-major) x Bw(NxK bf16 weight layout)^T [+ bias].
// Block = 8 waves = 128 rows x 64 cols. The 64-col B panel is shared by all
// waves: staged into LDS in KC-chunks via async-to-LDS, double buffered so the
// DMA of chunk c+1 overlaps the 16 WMMAs of chunk c. All four B fragments of a
// k-step are held in distinct registers so their ds_loads overlap WMMA issue.
__global__ __launch_bounds__(256) void gemm_bf16_f32(
        const __bf16* __restrict__ A, const __bf16* __restrict__ Bw,
        float* __restrict__ C, const float* __restrict__ bias,
        int M, int N, int K) {
    __shared__ __align__(16) __bf16 ldsB[2][64 * (KC + LDSPAD)];

    const int lane = threadIdx.x & 31;
    const int warp = threadIdx.x >> 5;
    const int nt64 = N >> 6;
    const int bn   = blockIdx.x % nt64;
    const int bm   = blockIdx.x / nt64;
    const int col0 = bn * 64;
    const int row0 = bm * 128 + warp * 16;

    // stage one KC-chunk of the 64xK B panel into LDS buffer `buf`
    auto stage = [&](int c, int buf) {
        #pragma unroll
        for (int it = 0; it < 4; ++it) {              // 256 thr x 4 x 16B = 16KB
            int l   = it * 256 + threadIdx.x;
            int row = l >> 4;                          // 0..63
            int seg = l & 15;                          // 16B segment within row
            const __bf16* g = Bw + (size_t)(col0 + row) * K + c * KC + seg * 8;
            __bf16* s = &ldsB[buf][row * (KC + LDSPAD) + seg * 8];
            ASYNC_COPY16(g, s);
        }
    };

    v8f acc[4] = {};
    const int nchunks = K / KC;
    int buf = 0;

    stage(0, 0);
    wait_async_copies();
    __syncthreads();

    for (int c = 0; c < nchunks; ++c) {
        if (c + 1 < nchunks) {
            stage(c + 1, buf ^ 1);
            __builtin_prefetch(A + (size_t)(row0 + (lane & 15)) * K + (c + 1) * KC, 0, 3);
        }
        #pragma unroll
        for (int kk = 0; kk < KC; kk += 32) {
            v16bf a = load_frag_a(A, K, row0, c * KC + kk, lane);
            v16bf bfr[4];
            #pragma unroll
            for (int j = 0; j < 4; ++j) {
                // B fragment from LDS (row stride KC+8 elems = 272B)
                int n  = lane & 15;
                int hi = (lane >> 4) & 1;
                const __bf16* p = &ldsB[buf][(j * 16 + n) * (KC + LDSPAD) + kk + hi * 16];
                union { v16bf v; uint4 q[2]; } u;
                u.q[0] = *(const uint4*)(p);
                u.q[1] = *(const uint4*)(p + 8);
                bfr[j] = u.v;
            }
            #pragma unroll
            for (int j = 0; j < 4; ++j)
                acc[j] = wmma_bf16(a, bfr[j], acc[j]);
        }
        wait_async_copies();
        __syncthreads();
        buf ^= 1;
    }

    int n = lane & 15, hi = lane >> 4;
    #pragma unroll
    for (int j = 0; j < 4; ++j) {
        #pragma unroll
        for (int r = 0; r < 8; ++r) {
            int row = row0 + r + hi * 8;
            int col = col0 + j * 16 + n;
            float v = acc[j][r];
            if (bias) v += bias[col];
            C[(size_t)row * N + col] = v;
        }
    }
}

// k-softmax over the sequence axis; writes kT[bh][d][i] (bf16, i contiguous).
// One 256-thread block per (b,h,d) column.
__global__ __launch_bounds__(256) void softmax_k_kernel(
        const float* __restrict__ qkv, __bf16* __restrict__ kT) {
    __shared__ float redm[8];
    __shared__ float reds[8];
    int idx = blockIdx.x;                 // (b*8+h)*64 + d, 0..1023
    int d = idx & 63, h = (idx >> 6) & 7, b = idx >> 9;
    const float* src = qkv + (size_t)(b * SEQ) * QKV3 + (DIM + h * DH + d);
    int tid = threadIdx.x, lane = tid & 31, w = tid >> 5;

    float vals[8];
    float m = -3.0e38f;
    #pragma unroll
    for (int t = 0; t < 8; ++t) {
        vals[t] = src[(size_t)(tid + t * 256) * QKV3];
        m = fmaxf(m, vals[t]);
    }
    m = wave_max(m);
    if (lane == 0) redm[w] = m;
    __syncthreads();
    m = redm[0];
    #pragma unroll
    for (int i = 1; i < 8; ++i) m = fmaxf(m, redm[i]);

    float s = 0.f;
    #pragma unroll
    for (int t = 0; t < 8; ++t) { vals[t] = __expf(vals[t] - m); s += vals[t]; }
    s = wave_sum(s);
    if (lane == 0) reds[w] = s;
    __syncthreads();
    s = 0.f;
    #pragma unroll
    for (int i = 0; i < 8; ++i) s += reds[i];
    float inv = 1.0f / s;

    __bf16* dst = kT + (size_t)idx * SEQ;
    #pragma unroll
    for (int t = 0; t < 8; ++t) dst[tid + t * 256] = (__bf16)(vals[t] * inv);
}

// q-softmax over head dim (64); writes qT[bh][e][i] bf16 (i contiguous) and
// converts the v slab to bf16 in (b,i,h,d) order. One wave per (b,h,i) row.
__global__ __launch_bounds__(256) void softmax_q_v_kernel(
        const float* __restrict__ qkv, __bf16* __restrict__ qT,
        __bf16* __restrict__ vb) {
    int lane = threadIdx.x & 31;
    int gw   = blockIdx.x * 8 + (threadIdx.x >> 5);   // 0 .. 32767
    int i = gw & (SEQ - 1);
    int h = (gw >> 11) & 7;
    int b = gw >> 14;

    const float* base = qkv + (size_t)(b * SEQ + i) * QKV3;
    float q0 = base[h * DH + lane];
    float q1 = base[h * DH + lane + 32];
    float m  = wave_max(fmaxf(q0, q1));
    q0 = __expf(q0 - m);
    q1 = __expf(q1 - m);
    float inv = 1.0f / wave_sum(q0 + q1);

    size_t o = (size_t)((b * HEADS + h) * DH) * SEQ + i;
    qT[o + (size_t)lane * SEQ]        = (__bf16)(q0 * inv);
    qT[o + (size_t)(lane + 32) * SEQ] = (__bf16)(q1 * inv);

    size_t vo = (size_t)(b * SEQ + i) * INNER + h * DH;
    vb[vo + lane]      = (__bf16)base[2 * DIM + h * DH + lane];
    vb[vo + lane + 32] = (__bf16)base[2 * DIM + h * DH + lane + 32];
}

// KQ[bh] = kT(64x2048) @ q(2048x64); both operands have contiguous K runs.
// Stores transposed as KQt[bh][e][d] bf16 so the next GEMM sees weight layout.
__global__ __launch_bounds__(256) void kq_gemm_kernel(
        const __bf16* __restrict__ kT, const __bf16* __restrict__ qT,
        __bf16* __restrict__ KQt) {
    int lane = threadIdx.x & 31;
    int gw   = blockIdx.x * 8 + (threadIdx.x >> 5);   // 0..63
    int bh = gw >> 2;
    int mt = gw & 3;
    const __bf16* Ak = kT + (size_t)bh * DH * SEQ;    // (64 x 2048)
    const __bf16* Bq = qT + (size_t)bh * DH * SEQ;    // weight layout (64 x 2048)

    v8f acc[4] = {};
    for (int k0 = 0; k0 < SEQ; k0 += 32) {
        v16bf a = load_frag_a(Ak, SEQ, mt * 16, k0, lane);
        v16bf bfr[4];
        #pragma unroll
        for (int j = 0; j < 4; ++j)
            bfr[j] = load_frag_b(Bq, SEQ, j * 16, k0, lane);
        #pragma unroll
        for (int j = 0; j < 4; ++j)
            acc[j] = wmma_bf16(a, bfr[j], acc[j]);
    }
    int n = lane & 15, hi = lane >> 4;
    #pragma unroll
    for (int j = 0; j < 4; ++j) {
        // KQt[bh][e][d]: e = j*16+n, d = mt*16 + hi*8 + r  (8 contiguous bf16)
        union { uint4 q; __bf16 h8[8]; } u;
        #pragma unroll
        for (int r = 0; r < 8; ++r) u.h8[r] = (__bf16)acc[j][r];
        size_t off = (size_t)bh * DH * DH + (size_t)(j * 16 + n) * DH + mt * 16 + hi * 8;
        *(uint4*)(KQt + off) = u.q;
    }
}

// H[b,i,h*64+e] = v[b,h](2048x64) @ KQ[bh](64x64). A straight from the bf16 v
// slab (row stride 512), B = KQt weight layout. Output bf16 for final GEMM.
__global__ __launch_bounds__(256) void av_gemm_kernel(
        const __bf16* __restrict__ vb, const __bf16* __restrict__ KQt,
        __bf16* __restrict__ Hb) {
    int lane = threadIdx.x & 31;
    int gw   = blockIdx.x * 8 + (threadIdx.x >> 5);   // 0..2047
    int bh = gw >> 7;
    int mt = gw & 127;
    int b = bh >> 3, h = bh & 7;
    const __bf16* Av = vb + (size_t)(b * SEQ) * INNER + h * DH;  // lda = 512
    const __bf16* Bk = KQt + (size_t)bh * DH * DH;               // (64 x 64)

    v8f acc[4] = {};
    #pragma unroll
    for (int k0 = 0; k0 < DH; k0 += 32) {
        v16bf a = load_frag_a(Av, INNER, mt * 16, k0, lane);
        v16bf bfr[4];
        #pragma unroll
        for (int j = 0; j < 4; ++j)
            bfr[j] = load_frag_b(Bk, DH, j * 16, k0, lane);
        #pragma unroll
        for (int j = 0; j < 4; ++j)
            acc[j] = wmma_bf16(a, bfr[j], acc[j]);
    }
    int n = lane & 15, hi = lane >> 4;
    #pragma unroll
    for (int j = 0; j < 4; ++j) {
        #pragma unroll
        for (int r = 0; r < 8; ++r) {
            int row = mt * 16 + r + hi * 8;
            int col = j * 16 + n;
            Hb[(size_t)(b * SEQ + row) * INNER + h * DH + col] = (__bf16)acc[j][r];
        }
    }
}

// ---------------------------------------------------------------------------

extern "C" void kernel_launch(void* const* d_in, const int* in_sizes, int n_in,
                              void* d_out, int out_size, void* d_ws, size_t ws_size,
                              hipStream_t stream) {
    const float* x     = (const float*)d_in[0];   // (2,2048,512)
    const float* w_qkv = (const float*)d_in[1];   // (1536,512)
    const float* w_out = (const float*)d_in[2];   // (512,512)
    const float* b_out = (const float*)d_in[3];   // (512,)
    float* out = (float*)d_out;

    char* ws = (char*)d_ws;
    size_t off = 0;
    auto take = [&](size_t bytes) -> char* {
        char* p = ws + off;
        off += (bytes + 255) & ~(size_t)255;
        return p;
    };
    __bf16* xb    = (__bf16*)take((size_t)ROWS * DIM   * 2);
    __bf16* wqkvb = (__bf16*)take((size_t)QKV3 * DIM   * 2);
    __bf16* woutb = (__bf16*)take((size_t)DIM  * INNER * 2);
    float*  qkvf  = (float*) take((size_t)ROWS * QKV3  * 4);
    __bf16* kT    = (__bf16*)take((size_t)BATCH * HEADS * DH * SEQ * 2);
    __bf16* qT    = (__bf16*)take((size_t)BATCH * HEADS * DH * SEQ * 2);
    __bf16* vb    = (__bf16*)take((size_t)ROWS * INNER * 2);
    __bf16* KQt   = (__bf16*)take((size_t)BATCH * HEADS * DH * DH * 2);
    __bf16* Hb    = (__bf16*)take((size_t)ROWS * INNER * 2);

    // 1) fp32 -> bf16 operand conversion
    cvt_f32_bf16<<<(ROWS * DIM + 255) / 256, 256, 0, stream>>>(x, xb, ROWS * DIM);
    cvt_f32_bf16<<<(QKV3 * DIM + 255) / 256, 256, 0, stream>>>(w_qkv, wqkvb, QKV3 * DIM);
    cvt_f32_bf16<<<(DIM * INNER + 255) / 256, 256, 0, stream>>>(w_out, woutb, DIM * INNER);

    // 2) qkv = x @ w_qkv^T  (4096x512x1536) -> f32
    gemm_bf16_f32<<<(ROWS / 128) * (QKV3 / 64), 256, 0, stream>>>(
        xb, wqkvb, qkvf, nullptr, ROWS, QKV3, DIM);

    // 3) softmaxes + v conversion
    softmax_k_kernel<<<BATCH * HEADS * DH, 256, 0, stream>>>(qkvf, kT);
    softmax_q_v_kernel<<<BATCH * HEADS * SEQ / 8, 256, 0, stream>>>(qkvf, qT, vb);

    // 4) KQ[bh] = kT @ q  (64x2048x64 per head) -> KQt bf16
    kq_gemm_kernel<<<8, 256, 0, stream>>>(kT, qT, KQt);

    // 5) H = v @ KQ per head  (2048x64x64) -> Hb bf16
    av_gemm_kernel<<<256, 256, 0, stream>>>(vb, KQt, Hb);

    // 6) out = H @ w_out^T + b_out  (4096x512x512) -> f32 d_out
    gemm_bf16_f32<<<(ROWS / 128) * (INNER / 64), 256, 0, stream>>>(
        Hb, woutb, out, b_out, ROWS, INNER, DIM);

    (void)in_sizes; (void)n_in; (void)out_size; (void)ws_size;
}
